// WitchAttention_53884659695646
// MI455X (gfx1250) — compile-verified, matrix operand order checked by reading, not run on hardware
//
#include <hip/hip_runtime.h>
#include <hip/hip_bf16.h>
#include <math.h>

typedef __bf16 bf16;
typedef __attribute__((ext_vector_type(16))) __bf16 v16bf;
typedef __attribute__((ext_vector_type(8)))  float   v8f;

#define H_   2048
#define NH_  16
#define NKV_ 4
#define D_   128
#define S_   2048
#define B_   2
#define MID_ 512
#define QS_  (NH_*D_)    // 2048
#define KVS_ (NKV_*D_)   // 512
#define NT_  (B_*S_)     // 4096 tokens

union FragA { v16bf v; uint4 q[2]; };

// Load a 16x32 bf16 fragment in the CDNA5 WMMA A/B lane layout from a
// row-major tile (ld = row stride in elements, multiple of 8).
// lanes 0-15: row = lane, K chunks {0..7, 16..23}
// lanes 16-31: row = lane-16, K chunks {8..15, 24..31}
__device__ inline v16bf load_frag_rowmajor(const bf16* __restrict__ base, int ld) {
  const int lane = threadIdx.x & 31;
  const int row  = lane & 15;
  const int koff = (lane < 16) ? 0 : 8;
  const bf16* p = base + (size_t)row * ld + koff;
  FragA f;
  f.q[0] = *reinterpret_cast<const uint4*>(p);
  f.q[1] = *reinterpret_cast<const uint4*>(p + 16);
  return f.v;
}

// ---------------------------------------------------------------- converts
__global__ void cvt_f32_bf16(const float* __restrict__ src, bf16* __restrict__ dst, int n) {
  int i = blockIdx.x * 256 + threadIdx.x;
  if (i < n) dst[i] = (bf16)src[i];
}

// W [K,N] f32 row-major -> Wt [N,K] bf16 row-major
__global__ void cvt_transpose(const float* __restrict__ W, bf16* __restrict__ Wt,
                              int K, int N) {
  long long i = (long long)blockIdx.x * 256 + threadIdx.x;
  if (i < (long long)K * N) {
    int n = (int)(i / K), k = (int)(i % K);
    Wt[i] = (bf16)W[(size_t)k * N + n];
  }
}

// ---------------------------------------------------------------- GEMM
// C[M,N] = A[M,K] * Bt[N,K]^T + bias[N];  A,Bt bf16 row-major, C f32.
// Block = 8 waves, tile 128(M) x 64(N); B tile staged in LDS, double-buffered.
// A fragments are wave-private rows -> direct global_load_b128.
// All 4 B fragments are pulled from LDS before the 4 WMMAs so a single
// s_wait_dscnt covers the batch and the WMMAs issue back-to-back.
__global__ void gemm_bf16_wmma(const bf16* __restrict__ A, const bf16* __restrict__ Bt,
                               const float* __restrict__ bias, float* __restrict__ C,
                               int M, int N, int K) {
  __shared__ __align__(16) bf16 bs[2][64 * 32];          // 2 x 4 KB
  const int wave = threadIdx.x >> 5;
  const int lane = threadIdx.x & 31;
  const int m0 = blockIdx.y * 128 + wave * 16;
  const int n0 = blockIdx.x * 64;

  // cooperative B-tile copy: 256 threads = 64 rows x 4 chunks of 16B
  const int crow = threadIdx.x >> 2;                     // 0..63
  const int cchk = (threadIdx.x & 3) * 8;                // elem offset 0,8,16,24
  const bf16* brow = Bt + (size_t)(n0 + crow) * K + cchk;

  const v8f vzero = {0.f,0.f,0.f,0.f,0.f,0.f,0.f,0.f};
  v8f acc[4] = {vzero, vzero, vzero, vzero};

  // prologue: fill buffer 0
  *reinterpret_cast<uint4*>(&bs[0][crow * 32 + cchk]) =
      *reinterpret_cast<const uint4*>(brow);
  __syncthreads();

  int cur = 0;
  for (int k0 = 0; k0 < K; k0 += 32) {
    uint4 bnext;
    const bool more = (k0 + 32) < K;
    if (more)            // overlap next tile's global loads with current WMMAs
      bnext = *reinterpret_cast<const uint4*>(brow + (k0 + 32));

    v16bf a = load_frag_rowmajor(A + (size_t)m0 * K + k0, K);
    v16bf bfr[4];
#pragma unroll
    for (int t = 0; t < 4; ++t)
      bfr[t] = load_frag_rowmajor(&bs[cur][(t * 16) * 32], 32);
#pragma unroll
    for (int t = 0; t < 4; ++t)
      acc[t] = __builtin_amdgcn_wmma_f32_16x16x32_bf16(false, a, false, bfr[t],
                                                       (short)0, acc[t], false, false);
    if (more) {
      __syncthreads();   // reads of bs[cur^1] are 2 barriers back -> safe to overwrite
      *reinterpret_cast<uint4*>(&bs[cur ^ 1][crow * 32 + cchk]) = bnext;
      __syncthreads();
      cur ^= 1;
    }
  }

  const int hi = (lane >= 16) ? 8 : 0;
  const int nc = lane & 15;
#pragma unroll
  for (int t = 0; t < 4; ++t) {
    const int col = n0 + t * 16 + nc;
    const float bcol = bias[col];
#pragma unroll
    for (int i = 0; i < 8; ++i)
      C[(size_t)(m0 + hi + i) * N + col] = acc[t][i] + bcol;
  }
}

// ---------------------------------------------------------------- RoPE + pack
// qkv f32 [B,S,2*QS] (q | q_gate), k/v f32 [B,S,KVS]
// -> Qb bf16 [B,NH,S,D], Kb bf16 [B,NKV,S,D], Vt bf16 [B,NKV,D,S]
__global__ void rope_pack(const float* __restrict__ qkv, const float* __restrict__ kbuf,
                          const float* __restrict__ vbuf, const float* __restrict__ cosb,
                          const float* __restrict__ sinb, bf16* __restrict__ Qb,
                          bf16* __restrict__ Kb, bf16* __restrict__ Vt) {
  const int d = threadIdx.x;       // 0..127
  const int s = blockIdx.x;
  const int b = blockIdx.y;
  const float c  = cosb[s * D_ + d];
  const float sn = sinb[s * D_ + d];

  const float* qrow = qkv + (size_t)(b * S_ + s) * (2 * QS_);
  for (int h = 0; h < NH_; ++h) {
    float x  = qrow[h * D_ + d];
    float xr = (d < D_ / 2) ? -qrow[h * D_ + d + D_ / 2] : qrow[h * D_ + d - D_ / 2];
    Qb[(((size_t)b * NH_ + h) * S_ + s) * D_ + d] = (bf16)(x * c + xr * sn);
  }
  const float* krow = kbuf + (size_t)(b * S_ + s) * KVS_;
  const float* vrow = vbuf + (size_t)(b * S_ + s) * KVS_;
  for (int g = 0; g < NKV_; ++g) {
    float x  = krow[g * D_ + d];
    float xr = (d < D_ / 2) ? -krow[g * D_ + d + D_ / 2] : krow[g * D_ + d - D_ / 2];
    Kb[(((size_t)b * NKV_ + g) * S_ + s) * D_ + d] = (bf16)(x * c + xr * sn);
    Vt[(((size_t)b * NKV_ + g) * D_ + d) * S_ + s] = (bf16)vrow[g * D_ + d];
  }
}

// ---------------------------------------------------------------- attention
// One wave per (b, head, 16-row Q tile). Flash-style over 32-col KV blocks.
__global__ void attn_wmma(const bf16* __restrict__ Qb, const bf16* __restrict__ Kb,
                          const bf16* __restrict__ Vt, const float* __restrict__ qkv,
                          bf16* __restrict__ Attn) {
  __shared__ __align__(16) bf16 pbuf[16 * 32];
  const int lane = threadIdx.x & 31;
  const int bh = blockIdx.x;
  const int b = bh / NH_, h = bh % NH_;
  const int g = h / (NH_ / NKV_);
  const int q0 = blockIdx.y * 16;
  const float scale = 0.08838834764831845f;   // 1/sqrt(128)

  v16bf qf[4];
#pragma unroll
  for (int kc = 0; kc < 4; ++kc)
    qf[kc] = load_frag_rowmajor(Qb + (((size_t)b * NH_ + h) * S_ + q0) * D_ + kc * 32, D_);

  const v8f vzero = {0.f,0.f,0.f,0.f,0.f,0.f,0.f,0.f};
  v8f o[8];
  float mrow[8], lrow[8];
#pragma unroll
  for (int t = 0; t < 8; ++t) o[t] = vzero;
#pragma unroll
  for (int i = 0; i < 8; ++i) { mrow[i] = -1e30f; lrow[i] = 0.f; }

  const int hi = (lane >= 16) ? 8 : 0;
  const int nc = lane & 15;

  for (int j0 = 0; j0 < q0 + 16; j0 += 32) {
    v8f sc0 = vzero, sc1 = vzero;
    const bf16* kbase = Kb + (((size_t)b * NKV_ + g) * S_ + j0) * D_;
    // preload all 8 K fragments, then run the 8 score WMMAs back-to-back
    v16bf kf[8];
#pragma unroll
    for (int kc = 0; kc < 4; ++kc) {
      kf[kc]     = load_frag_rowmajor(kbase + kc * 32, D_);
      kf[4 + kc] = load_frag_rowmajor(kbase + (size_t)16 * D_ + kc * 32, D_);
    }
#pragma unroll
    for (int kc = 0; kc < 4; ++kc) {
      sc0 = __builtin_amdgcn_wmma_f32_16x16x32_bf16(false, qf[kc], false, kf[kc],
                                                    (short)0, sc0, false, false);
      sc1 = __builtin_amdgcn_wmma_f32_16x16x32_bf16(false, qf[kc], false, kf[4 + kc],
                                                    (short)0, sc1, false, false);
    }
    // online softmax over these 32 columns (causal mask applied analytically)
#pragma unroll
    for (int i = 0; i < 8; ++i) {
      const int row = q0 + hi + i;
      const int c0 = j0 + nc, c1 = j0 + 16 + nc;
      float s0 = (c0 <= row) ? sc0[i] * scale : -1e30f;
      float s1 = (c1 <= row) ? sc1[i] * scale : -1e30f;
      float mx = fmaxf(s0, s1);
#pragma unroll
      for (int m = 8; m >= 1; m >>= 1) mx = fmaxf(mx, __shfl_xor(mx, m, 32));
      const float mnew = fmaxf(mrow[i], mx);
      float p0 = expf(s0 - mnew);
      float p1 = expf(s1 - mnew);
      float rs = p0 + p1;
#pragma unroll
      for (int m = 8; m >= 1; m >>= 1) rs += __shfl_xor(rs, m, 32);
      const float corr = expf(mrow[i] - mnew);
      lrow[i] = lrow[i] * corr + rs;
      mrow[i] = mnew;
#pragma unroll
      for (int t = 0; t < 8; ++t) o[t][i] *= corr;
      pbuf[(hi + i) * 32 + nc]      = (bf16)p0;
      pbuf[(hi + i) * 32 + 16 + nc] = (bf16)p1;
    }
    asm volatile("s_wait_dscnt 0" ::: "memory");   // LDS RAW before fragment reload
    v16bf pf = load_frag_rowmajor(pbuf, 32);
    // preload all 8 V fragments, then run the 8 P*V WMMAs back-to-back
    v16bf vf[8];
    const bf16* vcol = Vt + (((size_t)b * NKV_ + g) * D_) * S_ + j0;
#pragma unroll
    for (int t = 0; t < 8; ++t)
      vf[t] = load_frag_rowmajor(vcol + (size_t)(t * 16) * S_, S_);
#pragma unroll
    for (int t = 0; t < 8; ++t)
      o[t] = __builtin_amdgcn_wmma_f32_16x16x32_bf16(false, pf, false, vf[t],
                                                     (short)0, o[t], false, false);
  }
  // epilogue: normalize, sigmoid q-gate, store bf16 [B,S,QS]
#pragma unroll
  for (int t = 0; t < 8; ++t) {
#pragma unroll
    for (int i = 0; i < 8; ++i) {
      const int srow = q0 + hi + i;
      const int col = h * D_ + t * 16 + nc;
      const float gv = qkv[(size_t)(b * S_ + srow) * (2 * QS_) + QS_ + col];
      const float gate = 1.f / (1.f + expf(-gv));
      const float val = (o[t][i] / lrow[i]) * gate;
      Attn[(size_t)(b * S_ + srow) * QS_ + col] = (bf16)val;
    }
  }
}

// ---------------------------------------------------------------- conv gate
__global__ void conv1_gelu(const float* __restrict__ Out, const float* __restrict__ w1,
                           const float* __restrict__ b1, float* __restrict__ C1) {
  __shared__ float w1s[H_ * 5];                  // 40 KB
  const int mid = blockIdx.x, b = blockIdx.y;
  for (int i = threadIdx.x; i < H_ * 5; i += blockDim.x)
    w1s[i] = w1[(size_t)mid * H_ * 5 + i];
  __syncthreads();
  const float bv = b1[mid];
  for (int s = threadIdx.x; s < S_; s += blockDim.x) {
    float acc = bv;
#pragma unroll
    for (int t = 0; t < 5; ++t) {
      const int ss = s + t - 2;
      if (ss < 0 || ss >= S_) continue;
      const float* xrow = Out + (size_t)(b * S_ + ss) * H_;
      for (int hh = 0; hh < H_; ++hh)
        acc += xrow[hh] * w1s[hh * 5 + t];
    }
    C1[((size_t)(b * MID_ + mid)) * S_ + s] =
        0.5f * acc * (1.0f + erff(acc * 0.7071067811865475f));
  }
}

__global__ void conv2_gate(const float* __restrict__ C1, const float* __restrict__ w2,
                           const float* __restrict__ b2, const float* __restrict__ Out,
                           float* __restrict__ Y) {
  __shared__ float w2s[MID_ * 9];                // 18 KB
  const int hh = blockIdx.x, b = blockIdx.y;
  for (int i = threadIdx.x; i < MID_ * 9; i += blockDim.x)
    w2s[i] = w2[(size_t)hh * MID_ * 9 + i];
  __syncthreads();
  const float bv = b2[hh];
  for (int s = threadIdx.x; s < S_; s += blockDim.x) {
    float acc = bv;
    for (int m = 0; m < MID_; ++m) {
      const float* crow = C1 + ((size_t)(b * MID_ + m)) * S_;
#pragma unroll
      for (int t = 0; t < 9; ++t) {
        const int ss = s + t - 4;
        if (ss >= 0 && ss < S_) acc += crow[ss] * w2s[m * 9 + t];
      }
    }
    const float gate = 1.f / (1.f + expf(-acc));
    const size_t idx = (size_t)(b * S_ + s) * H_ + hh;
    Y[idx] = Out[idx] * gate;
  }
}

// ---------------------------------------------------------------- launch
extern "C" void kernel_launch(void* const* d_in, const int* in_sizes, int n_in,
                              void* d_out, int out_size, void* d_ws, size_t ws_size,
                              hipStream_t stream) {
  const float* hidden = (const float*)d_in[0];
  // d_in[1] = attention_mask (causal; applied analytically in attn_wmma)
  const float* rcos = (const float*)d_in[2];
  const float* rsin = (const float*)d_in[3];
  const float* Wq = (const float*)d_in[4];   const float* bq = (const float*)d_in[5];
  const float* Wk = (const float*)d_in[6];   const float* bk = (const float*)d_in[7];
  const float* Wv = (const float*)d_in[8];   const float* bv = (const float*)d_in[9];
  const float* Wo = (const float*)d_in[10];  const float* bo = (const float*)d_in[11];
  const float* w1 = (const float*)d_in[12];  const float* b1 = (const float*)d_in[13];
  const float* w2 = (const float*)d_in[14];  const float* b2 = (const float*)d_in[15];
  float* Y = (float*)d_out;

  char* ws = (char*)d_ws;
  size_t off = 0;
  auto wsalloc = [&](size_t bytes) -> char* {
    char* p = ws + off;
    off += (bytes + 255) & ~(size_t)255;
    return p;
  };
  bf16*  Xb   = (bf16*) wsalloc((size_t)NT_ * H_ * 2);
  bf16*  Wqt  = (bf16*) wsalloc((size_t)2 * QS_ * H_ * 2);   // [4096,2048]
  bf16*  Wkt  = (bf16*) wsalloc((size_t)KVS_ * H_ * 2);
  bf16*  Wvt  = (bf16*) wsalloc((size_t)KVS_ * H_ * 2);
  bf16*  Wot  = (bf16*) wsalloc((size_t)H_ * QS_ * 2);       // [2048,2048]
  float* tQKV = (float*)wsalloc((size_t)NT_ * 2 * QS_ * 4);
  float* tK   = (float*)wsalloc((size_t)NT_ * KVS_ * 4);
  float* tV   = (float*)wsalloc((size_t)NT_ * KVS_ * 4);
  bf16*  Qb   = (bf16*) wsalloc((size_t)B_ * NH_ * S_ * D_ * 2);
  bf16*  Kb   = (bf16*) wsalloc((size_t)B_ * NKV_ * S_ * D_ * 2);
  bf16*  Vtb  = (bf16*) wsalloc((size_t)B_ * NKV_ * D_ * S_ * 2);
  bf16*  Attn = (bf16*) wsalloc((size_t)NT_ * QS_ * 2);
  float* Out  = (float*)wsalloc((size_t)NT_ * H_ * 4);
  float* C1   = (float*)wsalloc((size_t)B_ * MID_ * S_ * 4);
  (void)ws_size; (void)in_sizes; (void)n_in; (void)out_size;

  // 1) bf16 conversions (weights transposed to [N,K] for fragment loads)
  { int n = NT_ * H_;
    cvt_f32_bf16<<<(n + 255) / 256, 256, 0, stream>>>(hidden, Xb, n); }
  { long long n = (long long)H_ * 2 * QS_;
    cvt_transpose<<<(unsigned)((n + 255) / 256), 256, 0, stream>>>(Wq, Wqt, H_, 2 * QS_); }
  { long long n = (long long)H_ * KVS_;
    cvt_transpose<<<(unsigned)((n + 255) / 256), 256, 0, stream>>>(Wk, Wkt, H_, KVS_);
    cvt_transpose<<<(unsigned)((n + 255) / 256), 256, 0, stream>>>(Wv, Wvt, H_, KVS_); }
  { long long n = (long long)QS_ * H_;
    cvt_transpose<<<(unsigned)((n + 255) / 256), 256, 0, stream>>>(Wo, Wot, QS_, H_); }

  // 2) projections (WMMA bf16, f32 accumulate, LDS double-buffered B tiles)
  gemm_bf16_wmma<<<dim3((2 * QS_) / 64, NT_ / 128), 256, 0, stream>>>(
      Xb, Wqt, bq, tQKV, NT_, 2 * QS_, H_);
  gemm_bf16_wmma<<<dim3(KVS_ / 64, NT_ / 128), 256, 0, stream>>>(
      Xb, Wkt, bk, tK, NT_, KVS_, H_);
  gemm_bf16_wmma<<<dim3(KVS_ / 64, NT_ / 128), 256, 0, stream>>>(
      Xb, Wvt, bv, tV, NT_, KVS_, H_);

  // 3) RoPE + repack (Q/K [b,h,s,d], V transposed [b,g,d,s])
  rope_pack<<<dim3(S_, B_), D_, 0, stream>>>(tQKV, tK, tV, rcos, rsin, Qb, Kb, Vtb);

  // 4) flash attention + sigmoid q-gate (WMMA)
  attn_wmma<<<dim3(B_ * NH_, S_ / 16), 32, 0, stream>>>(Qb, Kb, Vtb, tQKV, Attn);

  // 5) output projection (WMMA)
  gemm_bf16_wmma<<<dim3(H_ / 64, NT_ / 128), 256, 0, stream>>>(
      Attn, Wot, bo, Out, NT_, H_, QS_);

  // 6) conv token gate: conv1(5)+GELU -> conv2(9)+sigmoid, fused final multiply
  conv1_gelu<<<dim3(MID_, B_), 256, 0, stream>>>(Out, w1, b1, C1);
  conv2_gate<<<dim3(H_, B_), 256, 0, stream>>>(C1, w2, b2, Out, Y);
}
